// DenseFPModule_62989990363734
// MI455X (gfx1250) — matrix operand order repaired and checked by reading, not compiled, not run on hardware
//
#include <hip/hip_runtime.h>

// ---------------------------------------------------------------------------
// MI455X (gfx1250) DenseFPModule: 3-NN interpolate + concat + 2x(GEMM,BN,LReLU)
// GEMMs run as bf16 WMMA (v_wmma_f32_16x16x32_bf16) with fp32 accumulation.
// B tiles are staged block-wide in LDS via async global->LDS copies (ASYNCcnt)
// with a 2-deep double buffer, so the kt+1 tile copy overlaps the kt WMMAs.
// Operand tensors are pre-swizzled into A/B fragments in contiguous k order (k = 16*(lane/16) + e).
// ---------------------------------------------------------------------------

typedef __attribute__((ext_vector_type(16))) __bf16 v16bf;
typedef __attribute__((ext_vector_type(8)))  float  v8f;
typedef int v4i __attribute__((vector_size(16)));

#ifndef __has_builtin
#define __has_builtin(x) 0
#endif
#if __has_builtin(__builtin_amdgcn_global_load_async_to_lds_b128) && \
    __has_builtin(__builtin_amdgcn_s_wait_asynccnt)
#define USE_ASYNC_LDS 1
#else
#define USE_ASYNC_LDS 0
#endif

#define B_    8
#define N_    2048
#define M_    8192
#define C_    512
#define CS_   256
#define CIN_  768
#define H1_   512
#define H2_   256
#define KT1_  24          // 768 / 32
#define KT2_  16          // 512 / 32
#define MT_   512         // 8192 / 16
#define CNT_  65536.0f    // B*M  (BN population per channel)

__device__ __forceinline__ __bf16 f2bf(float f) {
  unsigned u = __float_as_uint(f);
  u += 0x7FFFu + ((u >> 16) & 1u);            // round-to-nearest-even
  unsigned short h = (unsigned short)(u >> 16);
  return __builtin_bit_cast(__bf16, h);
}
__device__ __forceinline__ float bf2f(__bf16 h) {
  unsigned short s = __builtin_bit_cast(unsigned short, h);
  return __uint_as_float(((unsigned)s) << 16);
}

// 16B global -> LDS stage (async when the toolchain exposes the CDNA5 builtin)
__device__ __forceinline__ void tile_copy16(const __bf16* g, __bf16* l) {
#if USE_ASYNC_LDS
  __builtin_amdgcn_global_load_async_to_lds_b128((v4i*)g, (v4i*)l, 0, 0);
#else
  *(uint4*)l = *(const uint4*)g;
#endif
}
__device__ __forceinline__ void tile_wait() {
#if USE_ASYNC_LDS
  __builtin_amdgcn_s_wait_asynccnt(0);
#endif
  __syncthreads();
}

// ---------------------------------------------------------------------------
// K0: zero the stats accumulators
// ---------------------------------------------------------------------------
__global__ void zero_kernel(float* __restrict__ p, int n) {
  int t = blockIdx.x * blockDim.x + threadIdx.x;
  if (t < n) p[t] = 0.0f;
}

// ---------------------------------------------------------------------------
// K1: 3-NN of each fine point among 2048 coarse points (refs staged in LDS)
// ---------------------------------------------------------------------------
__global__ __launch_bounds__(256) void three_nn_kernel(
    const float* __restrict__ pos, const float* __restrict__ pos_skip,
    int* __restrict__ idx, float* __restrict__ wgt) {
  __shared__ float rx[N_], ry[N_], rz[N_];
  const int b = blockIdx.y;
  const float* rp = pos + (size_t)b * N_ * 3;
  for (int t = threadIdx.x; t < N_; t += 256) {
    rx[t] = rp[t * 3 + 0];
    ry[t] = rp[t * 3 + 1];
    rz[t] = rp[t * 3 + 2];
  }
  __syncthreads();
  const int m = blockIdx.x * 256 + threadIdx.x;
  const float* qp = pos_skip + ((size_t)b * M_ + m) * 3;
  const float qx = qp[0], qy = qp[1], qz = qp[2];
  float d0 = 3.4e38f, d1 = 3.4e38f, d2 = 3.4e38f;
  int   i0 = 0,       i1 = 0,       i2 = 0;
  for (int n = 0; n < N_; ++n) {
    float dx = qx - rx[n], dy = qy - ry[n], dz = qz - rz[n];
    float d = dx * dx + dy * dy + dz * dz;
    if (d < d0)      { d2 = d1; i2 = i1; d1 = d0; i1 = i0; d0 = d; i0 = n; }
    else if (d < d1) { d2 = d1; i2 = i1; d1 = d;  i1 = n; }
    else if (d < d2) { d2 = d;  i2 = n; }
  }
  float r0 = 1.0f / (sqrtf(fmaxf(d0, 0.0f)) + 1e-8f);
  float r1 = 1.0f / (sqrtf(fmaxf(d1, 0.0f)) + 1e-8f);
  float r2 = 1.0f / (sqrtf(fmaxf(d2, 0.0f)) + 1e-8f);
  float inv = 1.0f / (r0 + r1 + r2);
  size_t o = ((size_t)b * M_ + m) * 3;
  idx[o + 0] = i0; idx[o + 1] = i1; idx[o + 2] = i2;
  wgt[o + 0] = r0 * inv; wgt[o + 1] = r1 * inv; wgt[o + 2] = r2 * inv;
}

// ---------------------------------------------------------------------------
// K2: convert a weight matrix W[O,I] (fp32) into bf16 A-fragment layout
// ---------------------------------------------------------------------------
__global__ void convert_w_kernel(const float* __restrict__ W,
                                 __bf16* __restrict__ Wb, int I, int total) {
  int t = blockIdx.x * 256 + threadIdx.x;
  if (t >= total) return;
  int e = t & 15, lane = (t >> 4) & 31, rest = t >> 9;
  int numKt = I >> 5;
  int kt = rest % numKt, ot = rest / numKt;
  int o = ot * 16 + (lane & 15);
  int k = kt * 32 + (lane & 16) + e;
  Wb[t] = f2bf(W[(size_t)o * I + k]);
}

// ---------------------------------------------------------------------------
// K3: build feats = [interp(x); x_skip] in bf16 B-fragment layout
// ---------------------------------------------------------------------------
__global__ void build_feats_kernel(const float* __restrict__ x,
                                   const float* __restrict__ xs,
                                   const int* __restrict__ idx,
                                   const float* __restrict__ wgt,
                                   __bf16* __restrict__ fb) {
  unsigned t = blockIdx.x * 256u + threadIdx.x;
  if (t >= (unsigned)(B_ * MT_ * KT1_ * 512)) return;
  int e = t & 15, lane = (t >> 4) & 31;
  unsigned t2 = t >> 9;
  int kt = (int)(t2 % KT1_);
  unsigned t3 = t2 / KT1_;
  int mt = (int)(t3 & (MT_ - 1));
  int b  = (int)(t3 >> 9);
  int i = kt * 32 + (lane & 16) + e;
  int m = mt * 16 + (lane & 15);
  float v;
  if (i < C_) {
    size_t o3 = ((size_t)b * M_ + m) * 3;
    const float* xr = x + ((size_t)b * C_ + i) * N_;
    v = wgt[o3 + 0] * xr[idx[o3 + 0]] +
        wgt[o3 + 1] * xr[idx[o3 + 1]] +
        wgt[o3 + 2] * xr[idx[o3 + 2]];
  } else {
    v = xs[((size_t)b * CS_ + (i - C_)) * M_ + m];
  }
  fb[t] = f2bf(v);
}

// ---------------------------------------------------------------------------
// K4: GEMM1  h1 = W1 x feats. Block = 8 waves = 128 O-rows x 64 m-cols.
// The shared 4KB B tile (4 fragments) is async-staged into an LDS double
// buffer; A fragments are rotated one iteration ahead from global (W1 lives
// in L2/WGP$). Epilogue: channel sum/sumsq + bf16 store in GEMM2's B layout.
// ---------------------------------------------------------------------------
__global__ __launch_bounds__(256) void gemm1_kernel(
    const __bf16* __restrict__ Ab, const __bf16* __restrict__ Bb,
    __bf16* __restrict__ Hb, float* __restrict__ sum, float* __restrict__ sq) {
  __shared__ __align__(16) __bf16 bbuf[2][4 * 512];
  const int lane = threadIdx.x & 31;
  const int wave = threadIdx.x >> 5;
  const int ot   = blockIdx.y * 8 + wave;   // 0..31 (O tile of 16 rows)
  const int mt0  = blockIdx.x * 4;          // 0..508
  const int b    = blockIdx.z;

  const __bf16* A     = Ab + (size_t)ot * (KT1_ * 512) + lane * 16;
  const __bf16* Bbase = Bb + (size_t)b * ((size_t)MT_ * KT1_ * 512) +
                        (size_t)mt0 * (KT1_ * 512);

  // copy lane mapping: thread -> (fragment j, 16B slice w)
  const int cj = threadIdx.x >> 6;          // 0..3
  const int cw = threadIdx.x & 63;          // 0..63
  const __bf16* gsrc = Bbase + (size_t)cj * (KT1_ * 512) + cw * 8;
  __bf16* ldst0 = &bbuf[0][cj * 512 + cw * 8];
  __bf16* ldst1 = &bbuf[1][cj * 512 + cw * 8];

  tile_copy16(gsrc, ldst0);                 // prologue: tile kt=0
  tile_wait();

  v8f acc[4] = {};
  v16bf ac = *(const v16bf*)(A);            // A fragment, kt=0
  for (int kt = 0; kt < KT1_; ++kt) {
    const int cur = kt & 1;
    if (kt + 1 < KT1_)                      // prefetch tile kt+1 (async)
      tile_copy16(gsrc + (size_t)(kt + 1) * 512, cur ? ldst0 : ldst1);
    const int ktn = (kt + 1 < KT1_) ? kt + 1 : kt;
    v16bf an = *(const v16bf*)(A + (size_t)ktn * 512);
    v16bf b0 = *(const v16bf*)&bbuf[cur][0 * 512 + lane * 16];
    v16bf b1 = *(const v16bf*)&bbuf[cur][1 * 512 + lane * 16];
    v16bf b2 = *(const v16bf*)&bbuf[cur][2 * 512 + lane * 16];
    v16bf b3 = *(const v16bf*)&bbuf[cur][3 * 512 + lane * 16];
    acc[0] = __builtin_amdgcn_wmma_f32_16x16x32_bf16(false, ac, false, b0, (short)0, acc[0], false, false);
    acc[1] = __builtin_amdgcn_wmma_f32_16x16x32_bf16(false, ac, false, b1, (short)0, acc[1], false, false);
    acc[2] = __builtin_amdgcn_wmma_f32_16x16x32_bf16(false, ac, false, b2, (short)0, acc[2], false, false);
    acc[3] = __builtin_amdgcn_wmma_f32_16x16x32_bf16(false, ac, false, b3, (short)0, acc[3], false, false);
    ac = an;
    if (kt + 1 < KT1_) tile_wait();         // tile kt+1 landed; bbuf[cur] free
  }

  const int col     = lane & 15;
  const int halfsel = (lane & 16) ? 8 : 0;  // C/D layout: M = r or r+8
  float s[8] = {}, q[8] = {};
  __bf16* outB = Hb + (size_t)b * ((size_t)MT_ * KT2_ * 512);
  #pragma unroll
  for (int j = 0; j < 4; ++j) {
    v8f d = acc[j];
    size_t mbase = (size_t)(mt0 + j) * (KT2_ * 512);
    #pragma unroll
    for (int r = 0; r < 8; ++r) {
      float v = d[r];
      s[r] += v; q[r] += v * v;
      int i2 = ot * 16 + r + halfsel;       // channel 0..511
      int kt2 = i2 >> 5, rem = i2 & 31;
      int lane2 = col + (rem & 16);
      int e2 = rem & 15;
      outB[mbase + (size_t)kt2 * 512 + lane2 * 16 + e2] = f2bf(v);
    }
  }
  #pragma unroll
  for (int r = 0; r < 8; ++r) {
    float sv = s[r], qv = q[r];
    for (int off = 8; off > 0; off >>= 1) {
      sv += __shfl_xor(sv, off, 16);
      qv += __shfl_xor(qv, off, 16);
    }
    if (col == 0) {
      int row = ot * 16 + r + halfsel;
      atomicAdd(&sum[row], sv);
      atomicAdd(&sq[row], qv);
    }
  }
}

// ---------------------------------------------------------------------------
// K5: BN stats -> per-channel affine (a = gamma*rsqrt(var+eps), c = beta - a*mu)
// ---------------------------------------------------------------------------
__global__ void finalize_bn_kernel(const float* __restrict__ sum,
                                   const float* __restrict__ sq,
                                   const float* __restrict__ gamma,
                                   const float* __restrict__ beta,
                                   float* __restrict__ a, float* __restrict__ c,
                                   int n) {
  int i = blockIdx.x * 256 + threadIdx.x;
  if (i >= n) return;
  float mu  = sum[i] * (1.0f / CNT_);
  float var = fmaxf(sq[i] * (1.0f / CNT_) - mu * mu, 0.0f);
  float inv = rsqrtf(var + 1e-5f);
  float ai = gamma[i] * inv;
  a[i] = ai;
  c[i] = beta[i] - mu * ai;
}

// ---------------------------------------------------------------------------
// K6: apply BN1 + LeakyReLU in place on the bf16 fragment-layout buffer.
// ---------------------------------------------------------------------------
__global__ void apply_bn1_kernel(__bf16* __restrict__ h,
                                 const float* __restrict__ a,
                                 const float* __restrict__ c) {
  unsigned t = blockIdx.x * 256u + threadIdx.x;
  if (t >= (unsigned)(B_ * MT_ * KT2_ * 512)) return;
  int e = t & 15, lane = (t >> 4) & 31, kt = (t >> 9) & 15;
  int i = kt * 32 + (lane & 16) + e;
  float v = bf2f(h[t]);
  v = a[i] * v + c[i];
  v = (v >= 0.0f) ? v : 0.2f * v;
  h[t] = f2bf(v);
}

// ---------------------------------------------------------------------------
// K7: GEMM2  h2 = W2 x h1' (same async-LDS pipelined structure). Writes
// pre-BN fp32 straight to d_out, accumulates layer-2 channel stats.
// ---------------------------------------------------------------------------
__global__ __launch_bounds__(256) void gemm2_kernel(
    const __bf16* __restrict__ Ab, const __bf16* __restrict__ Bb,
    float* __restrict__ out, float* __restrict__ sum, float* __restrict__ sq) {
  __shared__ __align__(16) __bf16 bbuf[2][4 * 512];
  const int lane = threadIdx.x & 31;
  const int wave = threadIdx.x >> 5;
  const int ot   = blockIdx.y * 8 + wave;   // 0..15
  const int mt0  = blockIdx.x * 4;
  const int b    = blockIdx.z;

  const __bf16* A     = Ab + (size_t)ot * (KT2_ * 512) + lane * 16;
  const __bf16* Bbase = Bb + (size_t)b * ((size_t)MT_ * KT2_ * 512) +
                        (size_t)mt0 * (KT2_ * 512);

  const int cj = threadIdx.x >> 6;
  const int cw = threadIdx.x & 63;
  const __bf16* gsrc = Bbase + (size_t)cj * (KT2_ * 512) + cw * 8;
  __bf16* ldst0 = &bbuf[0][cj * 512 + cw * 8];
  __bf16* ldst1 = &bbuf[1][cj * 512 + cw * 8];

  tile_copy16(gsrc, ldst0);
  tile_wait();

  v8f acc[4] = {};
  v16bf ac = *(const v16bf*)(A);
  for (int kt = 0; kt < KT2_; ++kt) {
    const int cur = kt & 1;
    if (kt + 1 < KT2_)
      tile_copy16(gsrc + (size_t)(kt + 1) * 512, cur ? ldst0 : ldst1);
    const int ktn = (kt + 1 < KT2_) ? kt + 1 : kt;
    v16bf an = *(const v16bf*)(A + (size_t)ktn * 512);
    v16bf b0 = *(const v16bf*)&bbuf[cur][0 * 512 + lane * 16];
    v16bf b1 = *(const v16bf*)&bbuf[cur][1 * 512 + lane * 16];
    v16bf b2 = *(const v16bf*)&bbuf[cur][2 * 512 + lane * 16];
    v16bf b3 = *(const v16bf*)&bbuf[cur][3 * 512 + lane * 16];
    acc[0] = __builtin_amdgcn_wmma_f32_16x16x32_bf16(false, ac, false, b0, (short)0, acc[0], false, false);
    acc[1] = __builtin_amdgcn_wmma_f32_16x16x32_bf16(false, ac, false, b1, (short)0, acc[1], false, false);
    acc[2] = __builtin_amdgcn_wmma_f32_16x16x32_bf16(false, ac, false, b2, (short)0, acc[2], false, false);
    acc[3] = __builtin_amdgcn_wmma_f32_16x16x32_bf16(false, ac, false, b3, (short)0, acc[3], false, false);
    ac = an;
    if (kt + 1 < KT2_) tile_wait();
  }

  const int col     = lane & 15;
  const int halfsel = (lane & 16) ? 8 : 0;
  float s[8] = {}, q[8] = {};
  float* outp = out + (size_t)b * H2_ * M_;
  #pragma unroll
  for (int j = 0; j < 4; ++j) {
    v8f d = acc[j];
    int m = (mt0 + j) * 16 + col;
    #pragma unroll
    for (int r = 0; r < 8; ++r) {
      float v = d[r];
      s[r] += v; q[r] += v * v;
      int row = ot * 16 + r + halfsel;      // channel 0..255
      outp[(size_t)row * M_ + m] = v;
    }
  }
  #pragma unroll
  for (int r = 0; r < 8; ++r) {
    float sv = s[r], qv = q[r];
    for (int off = 8; off > 0; off >>= 1) {
      sv += __shfl_xor(sv, off, 16);
      qv += __shfl_xor(qv, off, 16);
    }
    if (col == 0) {
      int row = ot * 16 + r + halfsel;
      atomicAdd(&sum[row], sv);
      atomicAdd(&sq[row], qv);
    }
  }
}

// ---------------------------------------------------------------------------
// K8: apply BN2 + LeakyReLU in place on d_out [B,256,M] fp32.
// ---------------------------------------------------------------------------
__global__ void apply_bn2_kernel(float* __restrict__ out,
                                 const float* __restrict__ a,
                                 const float* __restrict__ c) {
  unsigned t = blockIdx.x * 256u + threadIdx.x;
  if (t >= (unsigned)(B_ * H2_ * M_)) return;
  int i = (t >> 13) & 255;                  // M_ = 8192 = 1<<13
  float v = out[t];
  v = a[i] * v + c[i];
  out[t] = (v >= 0.0f) ? v : 0.2f * v;
}

// ---------------------------------------------------------------------------
extern "C" void kernel_launch(void* const* d_in, const int* in_sizes, int n_in,
                              void* d_out, int out_size, void* d_ws, size_t ws_size,
                              hipStream_t stream) {
  (void)in_sizes; (void)n_in; (void)out_size; (void)ws_size;
  const float* pos      = (const float*)d_in[0];
  const float* pos_skip = (const float*)d_in[1];
  const float* x        = (const float*)d_in[2];
  const float* x_skip   = (const float*)d_in[3];
  const float* W1       = (const float*)d_in[4];
  const float* g1       = (const float*)d_in[5];
  const float* be1      = (const float*)d_in[6];
  const float* W2       = (const float*)d_in[7];
  const float* g2       = (const float*)d_in[8];
  const float* be2      = (const float*)d_in[9];
  float* out = (float*)d_out;

  char* w = (char*)d_ws;
  size_t off = 0;
  auto take = [&](size_t bytes) -> char* {
    char* p = w + off;
    off = (off + bytes + 255) & ~(size_t)255;
    return p;
  };
  int*    idx    = (int*)   take((size_t)B_ * M_ * 3 * sizeof(int));
  float*  wgt    = (float*) take((size_t)B_ * M_ * 3 * sizeof(float));
  __bf16* w1b    = (__bf16*)take((size_t)H1_ * CIN_ * 2);
  __bf16* w2b    = (__bf16*)take((size_t)H2_ * H1_ * 2);
  __bf16* featsB = (__bf16*)take((size_t)B_ * CIN_ * M_ * 2);
  __bf16* h1b    = (__bf16*)take((size_t)B_ * H1_ * M_ * 2);
  float*  stats  = (float*) take(1536 * sizeof(float)); // sum1,sq1,sum2,sq2
  float*  coef   = (float*) take(1536 * sizeof(float)); // a1,c1,a2,c2
  float* sum1 = stats, *sq1 = stats + 512, *sum2 = stats + 1024, *sq2 = stats + 1280;
  float* a1 = coef, *c1 = coef + 512, *a2 = coef + 1024, *c2 = coef + 1280;

  zero_kernel<<<6, 256, 0, stream>>>(stats, 1536);

  three_nn_kernel<<<dim3(M_ / 256, B_), 256, 0, stream>>>(pos, pos_skip, idx, wgt);

  convert_w_kernel<<<(H1_ * CIN_) / 256, 256, 0, stream>>>(W1, w1b, CIN_, H1_ * CIN_);
  convert_w_kernel<<<(H2_ * H1_)  / 256, 256, 0, stream>>>(W2, w2b, H1_, H2_ * H1_);

  build_feats_kernel<<<(B_ * MT_ * KT1_ * 512) / 256, 256, 0, stream>>>(
      x, x_skip, idx, wgt, featsB);

  gemm1_kernel<<<dim3(MT_ / 4, H1_ / 128, B_), 256, 0, stream>>>(
      w1b, featsB, h1b, sum1, sq1);

  finalize_bn_kernel<<<2, 256, 0, stream>>>(sum1, sq1, g1, be1, a1, c1, H1_);

  apply_bn1_kernel<<<(B_ * MT_ * KT2_ * 512) / 256, 256, 0, stream>>>(h1b, a1, c1);

  gemm2_kernel<<<dim3(MT_ / 4, H2_ / 128, B_), 256, 0, stream>>>(
      w2b, h1b, out, sum2, sq2);

  finalize_bn_kernel<<<1, 256, 0, stream>>>(sum2, sq2, g2, be2, a2, c2, H2_);

  apply_bn2_kernel<<<(B_ * H2_ * M_) / 256, 256, 0, stream>>>(out, a2, c2);
}
